// BidirLSTM_53549652247167
// MI455X (gfx1250) — matrix operand order repaired and checked by reading, not compiled, not run on hardware
//
#include <hip/hip_runtime.h>
#include <hip/hip_bf16.h>

// ---------------------------------------------------------------------------
// Bidirectional 2-layer LSTM (B=64, T=1024, F=64, H=256) + Dense(1) head.
// Strategy: fused [h|x] @ [U;W] recurrent GEMM with v_wmma_f32_16x16x32_bf16,
// persistent workgroups per (batch-block, direction), wave-local gate math.
// ---------------------------------------------------------------------------

#define Bb   64
#define Tt   1024
#define Ff   64
#define Hh   256
#define NN   1024      // 4*H
#define LDSTRIDE 776   // halves per LDS row: 768 (H + max Kin) + 8 pad (bank spread)

typedef __attribute__((ext_vector_type(16))) __bf16 v16bf;
typedef __attribute__((ext_vector_type(8)))  float  v8f;

union BF16x16 { uint4 u[2]; v16bf v; };

__device__ __forceinline__ unsigned short f2bf(float f) {
    union { float f; unsigned int u; } x; x.f = f;
    unsigned int r = x.u + 0x7FFFu + ((x.u >> 16) & 1u);  // RNE
    return (unsigned short)(r >> 16);
}
__device__ __forceinline__ float bf2f(unsigned short h) {
    union { unsigned int u; float f; } x; x.u = ((unsigned int)h) << 16;
    return x.f;
}
__device__ __forceinline__ float sigf(float x) { return 1.0f / (1.0f + __expf(-x)); }
__device__ __forceinline__ float tanhfast(float x) { return 2.0f * sigf(2.0f * x) - 1.0f; }

// ---------------------------------------------------------------------------
// Pack Wcat = [U (H rows); Win (Kin rows)]  (both [rows, 1024] f32 row-major)
// into per-lane WMMA B-fragment order (bf16):
//   halfIndex = ((nt*kSteps + kk)*32 + lane)*16 + h
//   k = kk*32 + (lane>=16 ? 16 : 0) + h ,  n = nt*16 + (lane&15)
// ---------------------------------------------------------------------------
__global__ void lstm_pack_w(const float* __restrict__ U,
                            const float* __restrict__ Win,
                            int kSteps, unsigned short* __restrict__ out) {
    int idx = blockIdx.x * blockDim.x + threadIdx.x;
    int total = 64 * kSteps * 512;         // 64 N-tiles * kSteps * 32 lanes * 16 halves
    if (idx >= total) return;
    int h    = idx & 15;
    int lane = (idx >> 4) & 31;
    int rest = idx >> 9;                   // nt*kSteps + kk
    int kk   = rest % kSteps;
    int nt   = rest / kSteps;
    int k    = kk * 32 + ((lane >> 4) << 4) + h;
    int n    = nt * 16 + (lane & 15);
    float v  = (k < Hh) ? U[k * NN + n] : Win[(k - Hh) * NN + n];
    out[idx] = f2bf(v);
}

// ---------------------------------------------------------------------------
// Persistent recurrent scan over T for one layer, both directions.
// grid.x = 8  -> dir = bid&1, mblk = bid>>1 (batch rows [16*mblk, 16*mblk+16))
// block   = 512 threads = 16 waves; wave w owns hidden slice [16w, 16w+16).
// in  : layer0 -> f32 x [B,T,Kin] ; layer1 -> bf16 [B,T,Kin]
// out : bf16 [B,T,512], fwd in cols [0,256), bwd in cols [256,512)
// ---------------------------------------------------------------------------
__global__ __launch_bounds__(512)
void lstm_scan(const void* __restrict__ inPtr, int inIsBf16, int Kin, int kSteps,
               const unsigned short* __restrict__ packF,
               const unsigned short* __restrict__ packB,
               const float* __restrict__ biasF, const float* __restrict__ biasB,
               unsigned short* __restrict__ out) {
    __shared__ __align__(16) unsigned short shA[16 * LDSTRIDE];  // [row][ H h | Kin x ]

    const int tid  = threadIdx.x;
    const int lane = tid & 31;
    const int w    = tid >> 5;            // wave id 0..15
    const int dir  = blockIdx.x & 1;
    const int mblk = blockIdx.x >> 1;

    const unsigned short* pack = dir ? packB : packF;
    const float*          bias = dir ? biasB : biasF;

    // zero LDS (h must start at 0)
    for (int i = tid; i < 16 * LDSTRIDE; i += 512) shA[i] = 0;

    // per-lane constants
    const int ncol = 16 * w + (lane & 15);          // hidden unit this lane writes
    const float bI = bias[0 * Hh + ncol];
    const float bF = bias[1 * Hh + ncol];
    const float bG = bias[2 * Hh + ncol];
    const float bO = bias[3 * Hh + ncol];

    // B-fragment base offsets (halves) for this wave's 4 gate tiles
    const long tI = ((long)(w +  0) * kSteps) * 512 + (long)lane * 16;
    const long tF = ((long)(w + 16) * kSteps) * 512 + (long)lane * 16;
    const long tG = ((long)(w + 32) * kSteps) * 512 + (long)lane * 16;
    const long tO = ((long)(w + 48) * kSteps) * 512 + (long)lane * 16;

    v8f c = {};                                      // cell state [16x16], persistent

    const float*          inF = (const float*)inPtr;
    const unsigned short* inH = (const unsigned short*)inPtr;

    for (int t = 0; t < Tt; ++t) {
        const int tt = dir ? (Tt - 1 - t) : t;

        // ---- stage x_t into LDS cols [Hh, Hh+Kin) as bf16 ----
        if (inIsBf16) {
            for (int i = tid; i < 16 * Kin; i += 512) {
                int r = i / Kin, col = i - r * Kin;
                shA[r * LDSTRIDE + Hh + col] =
                    inH[((long)(mblk * 16 + r) * Tt + tt) * Kin + col];
            }
        } else {
            for (int i = tid; i < 16 * Kin; i += 512) {
                int r = i / Kin, col = i - r * Kin;
                shA[r * LDSTRIDE + Hh + col] =
                    f2bf(inF[((long)(mblk * 16 + r) * Tt + tt) * Kin + col]);
            }
        }
        __syncthreads();   // h (prev step) + x_t visible to all waves

        // ---- z = [h | x] @ Wcat via WMMA bf16 ----
        v8f zi = {}, zf = {}, zg = {}, zo = {};
        const int aoff = (lane & 15) * LDSTRIDE + ((lane >> 4) << 4);
        for (int kk = 0; kk < kSteps; ++kk) {
            BF16x16 a, bi, bfr, bg, bo;
            const unsigned short* ap = &shA[aoff + kk * 32];
            a.u[0] = *(const uint4*)(ap);          // K chunk 1 (8 halves)
            a.u[1] = *(const uint4*)(ap + 8);      // K chunk 2 (contiguous)

            const uint4* pI = (const uint4*)(pack + tI + (long)kk * 512);
            const uint4* pF = (const uint4*)(pack + tF + (long)kk * 512);
            const uint4* pG = (const uint4*)(pack + tG + (long)kk * 512);
            const uint4* pO = (const uint4*)(pack + tO + (long)kk * 512);
            bi.u[0]  = pI[0]; bi.u[1]  = pI[1];
            bfr.u[0] = pF[0]; bfr.u[1] = pF[1];
            bg.u[0]  = pG[0]; bg.u[1]  = pG[1];
            bo.u[0]  = pO[0]; bo.u[1]  = pO[1];

            zi = __builtin_amdgcn_wmma_f32_16x16x32_bf16(false, a.v, false, bi.v,  (short)0, zi, false, false);
            zf = __builtin_amdgcn_wmma_f32_16x16x32_bf16(false, a.v, false, bfr.v, (short)0, zf, false, false);
            zg = __builtin_amdgcn_wmma_f32_16x16x32_bf16(false, a.v, false, bg.v,  (short)0, zg, false, false);
            zo = __builtin_amdgcn_wmma_f32_16x16x32_bf16(false, a.v, false, bo.v,  (short)0, zo, false, false);
        }
        __syncthreads();   // all A reads done before h is overwritten

        // ---- gates, cell update, emit h (wave-local: same hidden slice) ----
        const int rbase = (lane >> 4) << 3;          // row offset 0 or 8
        const int gcol  = dir * Hh + ncol;           // column in [B,T,512] output
        #pragma unroll
        for (int e = 0; e < 8; ++e) {
            float iv = sigf(zi[e] + bI);
            float fv = sigf(zf[e] + bF);
            float gv = tanhfast(zg[e] + bG);
            float ov = sigf(zo[e] + bO);
            float cv = fv * c[e] + iv * gv;
            c[e] = cv;
            float hv = ov * tanhfast(cv);
            unsigned short hb = f2bf(hv);
            int row = e + rbase;
            shA[row * LDSTRIDE + ncol] = hb;                                   // next-step A
            out[((long)(mblk * 16 + row) * Tt + tt) * 512 + gcol] = hb;        // layer output
        }
        // next iteration's x-staging happens before the next __syncthreads()
    }
}

// ---------------------------------------------------------------------------
// Dense(1) head: out[b,t] = sum_d h1[b,t,d] * Wd[d] + bd   (h1 bf16, f32 out)
// ---------------------------------------------------------------------------
__global__ void lstm_dense(const unsigned short* __restrict__ h1,
                           const float* __restrict__ Wd,
                           const float* __restrict__ bd,
                           float* __restrict__ out) {
    int bt = blockIdx.x * blockDim.x + threadIdx.x;
    if (bt >= Bb * Tt) return;
    const uint4* hp = (const uint4*)(h1 + (long)bt * 512);
    float acc = bd[0];
    for (int i = 0; i < 64; ++i) {          // 64 * 8 halves = 512
        uint4 q = hp[i];
        unsigned int qq[4] = { q.x, q.y, q.z, q.w };
        #pragma unroll
        for (int j = 0; j < 4; ++j) {
            int d = i * 8 + j * 2;
            acc += bf2f((unsigned short)(qq[j] & 0xFFFFu))  * Wd[d];
            acc += bf2f((unsigned short)(qq[j] >> 16))      * Wd[d + 1];
        }
    }
    out[bt] = acc;
}

// ---------------------------------------------------------------------------
extern "C" void kernel_launch(void* const* d_in, const int* in_sizes, int n_in,
                              void* d_out, int out_size, void* d_ws, size_t ws_size,
                              hipStream_t stream) {
    (void)in_sizes; (void)n_in; (void)out_size; (void)ws_size;
    const float* x   = (const float*)d_in[0];
    const float* W0f = (const float*)d_in[1];
    const float* U0f = (const float*)d_in[2];
    const float* b0f = (const float*)d_in[3];
    const float* W0b = (const float*)d_in[4];
    const float* U0b = (const float*)d_in[5];
    const float* b0b = (const float*)d_in[6];
    const float* W1f = (const float*)d_in[7];
    const float* U1f = (const float*)d_in[8];
    const float* b1f = (const float*)d_in[9];
    const float* W1b = (const float*)d_in[10];
    const float* U1b = (const float*)d_in[11];
    const float* b1b = (const float*)d_in[12];
    const float* Wd  = (const float*)d_in[13];
    const float* bd  = (const float*)d_in[14];
    float* y = (float*)d_out;

    // ---- workspace carve-up (256B aligned) ----
    char* ws = (char*)d_ws;
    size_t cur = 0;
    auto alloc = [&](size_t bytes) -> char* {
        char* p = ws + cur;
        cur += (bytes + 255) & ~(size_t)255;
        return p;
    };
    const size_t p0Halves = (size_t)64 * 10 * 512;    // 327,680
    const size_t p1Halves = (size_t)64 * 24 * 512;    // 786,432
    const size_t hHalves  = (size_t)Bb * Tt * 512;    // 33,554,432
    unsigned short* p0f  = (unsigned short*)alloc(p0Halves * 2);
    unsigned short* p0b  = (unsigned short*)alloc(p0Halves * 2);
    unsigned short* p1f  = (unsigned short*)alloc(p1Halves * 2);
    unsigned short* p1b  = (unsigned short*)alloc(p1Halves * 2);
    unsigned short* out0 = (unsigned short*)alloc(hHalves * 2);
    unsigned short* out1 = (unsigned short*)alloc(hHalves * 2);

    // ---- 1) pack weights into WMMA B-fragment order (bf16) ----
    {
        int g0 = (int)((p0Halves + 255) / 256);
        int g1 = (int)((p1Halves + 255) / 256);
        lstm_pack_w<<<g0, 256, 0, stream>>>(U0f, W0f, 10, p0f);
        lstm_pack_w<<<g0, 256, 0, stream>>>(U0b, W0b, 10, p0b);
        lstm_pack_w<<<g1, 256, 0, stream>>>(U1f, W1f, 24, p1f);
        lstm_pack_w<<<g1, 256, 0, stream>>>(U1b, W1b, 24, p1b);
    }

    // ---- 2) layer 0 bidirectional scan (f32 input, Kin=64, K=320) ----
    lstm_scan<<<8, 512, 0, stream>>>(x, 0, Ff, 10, p0f, p0b, b0f, b0b, out0);

    // ---- 3) layer 1 bidirectional scan (bf16 input, Kin=512, K=768) ----
    lstm_scan<<<8, 512, 0, stream>>>(out0, 1, 512, 24, p1f, p1b, b1f, b1b, out1);

    // ---- 4) dense head ----
    lstm_dense<<<(Bb * Tt) / 256, 256, 0, stream>>>(out1, Wd, bd, y);
}
